// PotentialModel_39840116637788
// MI455X (gfx1250) — matrix operand order repaired and checked by reading, not compiled, not run on hardware
//
#include <hip/hip_runtime.h>
#include <hip/hip_bf16.h>

typedef _Float16 f16;
typedef __attribute__((ext_vector_type(16))) _Float16 v16h;
typedef __attribute__((ext_vector_type(8)))  float    v8f;

#define NEG_SLOPE 0.2f

// ---------------- problem constants ----------------
#define NN   16384
#define BB   16
#define EE   262144
#define IND  20
#define HH   3
#define DD   100
#define HD   300
#define NPG  (NN / BB)     // 1024 = 2^10
#define EPG  (EE / BB)     // 16384 = 2^14

// LDS strides (halfs). ACT_STR: 260 dwords == 4 mod 64 banks -> 16 A-fragment
// rows hit 16 distinct banks. WB_STR: 20 dwords -> 16 consecutive columns hit
// 16 distinct banks; fragments are 2x aligned b128 loads.
#define ACT_STR 520
#define WB_STR  40

__device__ __forceinline__ float leaky(float v) { return v > 0.f ? v : NEG_SLOPE * v; }

__device__ __forceinline__ unsigned encf(float f) {
    unsigned u = __float_as_uint(f);
    return (u & 0x80000000u) ? ~u : (u | 0x80000000u);
}
__device__ __forceinline__ float decf(unsigned u) {
    return (u & 0x80000000u) ? __uint_as_float(u & 0x7FFFFFFFu) : __uint_as_float(~u);
}

__global__ void k_zero(unsigned* p, long n) {
    long i = (long)blockIdx.x * blockDim.x + threadIdx.x;
    if (i < n) p[i] = 0u;
}

// ---------------- compile-time MLP configurations ----------------
// CFG 0: u2e GEMM 20->300      CFG 1: GAT GEMM 300->300
// CFG 2: energy head           CFG 3: force head       CFG 4: stress head
template<int CFG> struct Cfg;
template<> struct Cfg<0> { static constexpr int NL=1, INM=0, OUTM=3, ACTB=0;
                           static constexpr int dims[] = {IND, HD}; };
template<> struct Cfg<1> { static constexpr int NL=1, INM=0, OUTM=3, ACTB=0;
                           static constexpr int dims[] = {HD, HD}; };
template<> struct Cfg<2> { static constexpr int NL=6, INM=0, OUTM=0, ACTB=3;
                           static constexpr int dims[] = {300,512,256,128,64,32,1}; };
template<> struct Cfg<3> { static constexpr int NL=6, INM=1, OUTM=1, ACTB=3;
                           static constexpr int dims[] = {300,512,256,128,64,32,3}; };
template<> struct Cfg<4> { static constexpr int NL=6, INM=2, OUTM=2, ACTB=3;
                           static constexpr int dims[] = {300,512,256,128,64,32,6}; };

struct RT {
    const float* W[6];
    const float* b[6];
    const float* Xin;
    const float* atomCode;
    const int*   src;
    const int*   dst;
    const float* dist;
    const float* direction;
    float* out;
};

// ---- one 16x16 n-tile: unconditional 2x16B B load (zeros pre-staged), WMMA.
// All guards fold to constants except Nd=300 / out-dim layers (uniform branch).
#define FRAG(I, C0)                                                            \
    if constexpr (nTiles > (I) * 4) {                                          \
        const int nt = wave_n + ((I) << 2);                                    \
        if (((I) * 4 + 3 < nTiles) || (nt < nTiles)) {                         \
            const f16* bb = &wB[(nt * 16 + lr - (C0)) * WB_STR + hiHalf * 8]; \
            v16h bfr;                                                          \
            _Pragma("unroll")                                                  \
            for (int j = 0; j < 16; ++j) bfr[j] = bb[j + (j & 8)];             \
            acc[(I)] = __builtin_amdgcn_wmma_f32_16x16x32_f16(                 \
                false, afr, false, bfr, (short)0, acc[(I)], false, false);     \
        }                                                                      \
    }

#define WBF(I)                                                                 \
    if constexpr (nTiles > (I) * 4) {                                          \
        const int nt = wave_n + ((I) << 2);                                    \
        if (((I) * 4 + 3 < nTiles) || (nt < nTiles)) {                         \
            const int col = nt * 16 + lr;                                      \
            if ((Nd % 16 == 0) || (col < Nd)) {                                \
                const float bv = bias ? bias[col] : 0.f;                       \
                _Pragma("unroll")                                              \
                for (int r = 0; r < 8; ++r) {                                  \
                    const int row = wave_m * 16 + r + hiHalf * 8;              \
                    float v = acc[(I)][r] + bv;                                \
                    if constexpr (doAct) v = leaky(v);                         \
                    actA[row * ACT_STR + col] = (f16)v;                        \
                }                                                              \
            }                                                                  \
        }                                                                      \
    }

#define EPI(I)                                                                 \
    if constexpr (nTiles > (I) * 4) {                                          \
        const int nt = wave_n + ((I) << 2);                                    \
        if (((I) * 4 + 3 < nTiles) || (nt < nTiles)) {                         \
            const int col = nt * 16 + lr;                                      \
            if ((Nd % 16 == 0) || (col < Nd)) {                                \
                const float bv = bias ? bias[col] : 0.f;                       \
                _Pragma("unroll")                                              \
                for (int r = 0; r < 8; ++r) {                                  \
                    const int row = wave_m * 16 + r + hiHalf * 8;              \
                    const long g = rowBase + row;                              \
                    float v = acc[(I)][r] + bv;                                \
                    if constexpr (doAct) v = leaky(v);                         \
                    if constexpr (C::OUTM == 3) {                              \
                        p.out[g * (long)Nd + col] = v;                         \
                    } else if constexpr (C::OUTM == 0) {                       \
                        atomicAdd(&p.out[g >> 10], v * (1.0f / NPG));          \
                    } else if constexpr (C::OUTM == 1) {                       \
                        const int t = p.dst[g];                                \
                        atomicAdd(&p.out[(long)t * 3 + col],                   \
                                  v * p.direction[g * 3 + col]);               \
                    } else {                                                   \
                        atomicAdd(&p.out[(g >> 14) * 6 + col],                 \
                                  v * p.direction[g * 3 + (col % 3)] * (1.0f / EPG)); \
                    }                                                          \
                }                                                              \
            }                                                                  \
        }                                                                      \
    }

template<class C, int LI>
__device__ __forceinline__ void do_layer(const RT& p, f16* actA, f16* wB, int tid,
                                         int wave_m, int wave_n, int lr, int hiHalf,
                                         long rowBase) {
    constexpr int  Kd      = C::dims[LI];
    constexpr int  Nd      = C::dims[LI + 1];
    constexpr int  nTiles  = (Nd + 15) / 16;
    constexpr int  kSteps  = (Kd + 31) / 32;
    constexpr bool isLast  = (LI + 1 == C::NL);
    constexpr bool doAct   = (LI < C::ACTB);
    const float* W    = p.W[LI];
    const float* bias = p.b[LI];

    v8f acc[8];
    #pragma unroll
    for (int i = 0; i < 8; ++i) acc[i] = (v8f){0.f,0.f,0.f,0.f,0.f,0.f,0.f,0.f};

    for (int ks = 0; ks < kSteps; ++ks) {
        const int k0 = ks * 32;

        // ---- phase 0: columns [0, min(nTiles*16,256)) ----
        __syncthreads();   // prior wB reads + prior-layer actA writeback done
        {
            constexpr int colsPad0 = (nTiles * 16 < 256) ? nTiles * 16 : 256;
            if (tid < colsPad0) {
                const bool cOK = (colsPad0 <= Nd) || (tid < Nd);
                const float* wp = W + (long)k0 * Nd + tid;
                if (ks + 1 < kSteps) __builtin_prefetch(wp + (long)32 * Nd, 0, 1);
                f16* dp = &wB[tid * WB_STR];
                #pragma unroll 4
                for (int kk = 0; kk < 32; ++kk) {
                    const bool kOK = (Kd % 32 == 0) || (k0 + kk < Kd);
                    dp[kk] = (cOK && kOK) ? (f16)wp[(long)kk * Nd] : (f16)0.f;
                }
            }
        }
        __syncthreads();

        // A fragment (after barrier -> safe vs prior-layer writeback)
        v16h afr;
        {
            const f16* ab = &actA[(wave_m * 16 + lr) * ACT_STR + k0 + hiHalf * 8];
            #pragma unroll
            for (int j = 0; j < 8; ++j) afr[j] = ab[j];
            #pragma unroll
            for (int j = 0; j < 8; ++j) afr[8 + j] = ab[16 + j];
        }
        FRAG(0, 0) FRAG(1, 0) FRAG(2, 0) FRAG(3, 0)

        // ---- phase 1: columns [256, nTiles*16) ----
        if constexpr (nTiles * 16 > 256) {
            __syncthreads();
            {
                constexpr int colsPad1 = nTiles * 16 - 256;
                if (tid < colsPad1) {
                    const int col = 256 + tid;
                    const bool cOK = (256 + colsPad1 <= Nd) || (col < Nd);
                    const float* wp = W + (long)k0 * Nd + col;
                    f16* dp = &wB[tid * WB_STR];
                    #pragma unroll 4
                    for (int kk = 0; kk < 32; ++kk) {
                        const bool kOK = (Kd % 32 == 0) || (k0 + kk < Kd);
                        dp[kk] = (cOK && kOK) ? (f16)wp[(long)kk * Nd] : (f16)0.f;
                    }
                }
            }
            __syncthreads();
            FRAG(4, 256) FRAG(5, 256) FRAG(6, 256) FRAG(7, 256)
        }
    }
    __syncthreads();   // all actA reads for this layer complete

    if constexpr (!isLast) {
        WBF(0) WBF(1) WBF(2) WBF(3) WBF(4) WBF(5) WBF(6) WBF(7)
        // next layer's leading barrier orders these writes vs its reads
    } else {
        EPI(0) EPI(1) EPI(2) EPI(3) EPI(4) EPI(5) EPI(6) EPI(7)
    }
}

template<class C, int LI>
__device__ __forceinline__ void chain(const RT& p, f16* actA, f16* wB, int tid,
                                      int wave_m, int wave_n, int lr, int hiHalf,
                                      long rowBase) {
    do_layer<C, LI>(p, actA, wB, tid, wave_m, wave_n, lr, hiHalf, rowBase);
    if constexpr (LI + 1 < C::NL)
        chain<C, LI + 1>(p, actA, wB, tid, wave_m, wave_n, lr, hiHalf, rowBase);
}

template<int CFG>
__global__ __launch_bounds__(256) void k_mlp(RT p) {
    using C = Cfg<CFG>;
    __shared__ f16 actA[32 * ACT_STR];   // 33280 B
    __shared__ f16 wB[256 * WB_STR];     // 20480 B

    const int tid    = threadIdx.x;
    const int lane   = tid & 31;
    const int waveId = tid >> 5;
    const int wave_m = waveId & 1;
    const int wave_n = waveId >> 1;
    const int lr     = lane & 15;
    const int hiHalf = lane >> 4;
    const long rowBase = (long)blockIdx.x * 32;

    for (int i = tid; i < 32 * ACT_STR; i += 256) actA[i] = (f16)0.f;
    __syncthreads();

    constexpr int K0 = C::dims[0];
    if constexpr (C::INM == 0) {
        for (int r = 0; r < 32; ++r) {
            const float* xp = p.Xin + (rowBase + r) * K0;
            for (int c = tid; c < K0; c += 256)
                actA[r * ACT_STR + c] = (f16)xp[c];
        }
    } else {
        for (int r = 0; r < 32; ++r) {
            const long g = rowBase + r;
            const int s = p.src[g], t = p.dst[g];
            float dd = p.dist[g]; dd = dd < 0.01f ? 0.01f : dd;
            const float invd = 1.0f / dd;
            const float* hs = p.Xin + (long)s * HD;
            const float* ht = p.Xin + (long)t * HD;
            const float* ac = p.atomCode + (long)s * HD;   // used only if INM==2
            for (int c = tid; c < HD; c += 256) {
                int hh = c / DD;
                float mul = (hh == 0) ? invd : (hh == 1 ? dd : 1.0f);
                float v = (hs[c] + ht[c]) * mul;
                if constexpr (C::INM == 2) v += ac[c];
                actA[r * ACT_STR + c] = (f16)v;
            }
        }
    }

    chain<C, 0>(p, actA, wB, tid, wave_m, wave_n, lr, hiHalf, rowBase);
}

// ---------------- GAT attention kernels ----------------
__global__ void k_att_scores(const float* feat, const float* asrc, const float* adst,
                             float* es, float* ed) {
    int idx = blockIdx.x * blockDim.x + threadIdx.x;
    if (idx >= NN * HH) return;
    int n = idx / HH, hh = idx - n * HH;
    const float* f = feat + (long)n * HD + hh * DD;
    const float* a = asrc + hh * DD;
    const float* b = adst + hh * DD;
    float s = 0.f, d = 0.f;
    #pragma unroll 4
    for (int i = 0; i < DD; ++i) { float fv = f[i]; s += fv * a[i]; d += fv * b[i]; }
    es[idx] = s; ed[idx] = d;
}

__global__ void k_edge_logits(const float* es, const float* ed, const int* src,
                              const int* dst, float* elog, unsigned* menc) {
    long idx = (long)blockIdx.x * blockDim.x + threadIdx.x;
    if (idx >= (long)EE * HH) return;
    long e = idx / HH; int hh = (int)(idx - e * HH);
    float v = es[(long)src[e] * HH + hh] + ed[(long)dst[e] * HH + hh];
    v = leaky(v);
    elog[idx] = v;
    atomicMax(&menc[(long)dst[e] * HH + hh], encf(v));
}

__global__ void k_edge_expsum(float* elog, const unsigned* menc, float* ssum,
                              const int* dst) {
    long idx = (long)blockIdx.x * blockDim.x + threadIdx.x;
    if (idx >= (long)EE * HH) return;
    long e = idx / HH; int hh = (int)(idx - e * HH);
    float m = decf(menc[(long)dst[e] * HH + hh]);
    float ex = __expf(elog[idx] - m);
    elog[idx] = ex;
    atomicAdd(&ssum[(long)dst[e] * HH + hh], ex);
}

__global__ void k_aggregate(const float* feat, const float* ex, const float* ssum,
                            const int* src, const int* dst, const float* dist,
                            float* agg) {
    long idx = (long)blockIdx.x * blockDim.x + threadIdx.x;
    if (idx >= (long)EE * HD) return;
    long e = idx / HD; int c = (int)(idx - e * HD);
    int hh = c / DD;
    int t = dst[e];
    float dd = dist[e]; dd = dd < 0.01f ? 0.01f : dd;
    float dh = (hh == 0) ? dd : (hh == 1 ? 1.0f / dd : 1.0f);
    float a = ex[e * HH + hh] / ssum[(long)t * HH + hh] * dh;
    atomicAdd(&agg[(long)t * HD + c], feat[(long)src[e] * HD + c] * a);
}

__global__ void k_leaky(const float* in, float* out, long n) {
    long i = (long)blockIdx.x * blockDim.x + threadIdx.x;
    if (i < n) out[i] = leaky(in[i]);
}

// ---------------- host side ----------------
static inline int gup(long n, int b) { return (int)((n + b - 1) / b); }

extern "C" void kernel_launch(void* const* d_in, const int* in_sizes, int n_in,
                              void* d_out, int out_size, void* d_ws, size_t ws_size,
                              hipStream_t stream) {
    const float* x    = (const float*)d_in[0];
    const float* dist = (const float*)d_in[1];
    const float* dir  = (const float*)d_in[2];
    const int*   src  = (const int*)d_in[3];
    const int*   dst  = (const int*)d_in[4];
    const float* gatW[3] = {(const float*)d_in[5],  (const float*)d_in[6],  (const float*)d_in[7]};
    const float* gatB[3] = {(const float*)d_in[8],  (const float*)d_in[9],  (const float*)d_in[10]};
    const float* attS[3] = {(const float*)d_in[11], (const float*)d_in[12], (const float*)d_in[13]};
    const float* attD[3] = {(const float*)d_in[14], (const float*)d_in[15], (const float*)d_in[16]};
    const float* eW[6]; const float* eB[6];
    const float* fW[6]; const float* fB[6];
    const float* sW[6]; const float* sB[6];
    for (int i = 0; i < 6; ++i) {
        eW[i] = (const float*)d_in[17 + i];  eB[i] = (const float*)d_in[23 + i];
        fW[i] = (const float*)d_in[29 + i];  fB[i] = (const float*)d_in[35 + i];
        sW[i] = (const float*)d_in[41 + i];  sB[i] = (const float*)d_in[47 + i];
    }
    const float* u2eW = (const float*)d_in[53];

    float* ws = (float*)d_ws;
    float*    hbuf  = ws;
    float*    feat  = hbuf  + (long)NN * HD;
    float*    agg   = feat  + (long)NN * HD;
    float*    atomc = agg   + (long)NN * HD;
    float*    es    = atomc + (long)NN * HD;
    float*    ed    = es    + (long)NN * HH;
    unsigned* menc  = (unsigned*)(ed + (long)NN * HH);
    float*    ssum  = (float*)menc + (long)NN * HH;
    float*    elog  = ssum  + (long)NN * HH;

    float* outE = (float*)d_out;
    float* outF = outE + BB;
    float* outS = outF + (long)NN * 3;

    k_zero<<<gup(49264, 256), 256, 0, stream>>>((unsigned*)d_out, 49264);

    // atom_code = x @ u2e_W (no bias)
    {
        RT p{};
        p.W[0] = u2eW; p.b[0] = nullptr;
        p.Xin = x; p.out = atomc;
        k_mlp<0><<<NN / 32, 256, 0, stream>>>(p);
    }

    // ---- 3 GAT layers ----
    for (int l = 0; l < 3; ++l) {
        RT p{};
        p.W[0] = gatW[l]; p.b[0] = gatB[l];
        p.Xin = (l == 0) ? x : hbuf; p.out = feat;
        if (l == 0) k_mlp<0><<<NN / 32, 256, 0, stream>>>(
            RT{{gatW[0]}, {gatB[0]}, x, nullptr, nullptr, nullptr, nullptr, nullptr, feat});
        else        k_mlp<1><<<NN / 32, 256, 0, stream>>>(p);

        k_zero<<<gup((long)NN * HD, 256), 256, 0, stream>>>((unsigned*)agg, (long)NN * HD);
        k_zero<<<gup((long)NN * HH * 2, 256), 256, 0, stream>>>((unsigned*)menc, (long)NN * HH * 2);

        k_att_scores<<<gup((long)NN * HH, 256), 256, 0, stream>>>(feat, attS[l], attD[l], es, ed);
        k_edge_logits<<<gup((long)EE * HH, 256), 256, 0, stream>>>(es, ed, src, dst, elog, menc);
        k_edge_expsum<<<gup((long)EE * HH, 256), 256, 0, stream>>>(elog, menc, ssum, dst);
        k_aggregate<<<gup((long)EE * HD, 256), 256, 0, stream>>>(feat, elog, ssum, src, dst, dist, agg);
        k_leaky<<<gup((long)NN * HD, 256), 256, 0, stream>>>(agg, hbuf, (long)NN * HD);
    }

    // ---- energy head ----
    {
        RT p{};
        for (int i = 0; i < 6; ++i) { p.W[i] = eW[i]; p.b[i] = eB[i]; }
        p.Xin = hbuf; p.out = outE;
        k_mlp<2><<<NN / 32, 256, 0, stream>>>(p);
    }

    // ---- force head ----
    {
        RT p{};
        for (int i = 0; i < 6; ++i) { p.W[i] = fW[i]; p.b[i] = fB[i]; }
        p.Xin = hbuf;
        p.src = src; p.dst = dst; p.dist = dist; p.direction = dir;
        p.out = outF;
        k_mlp<3><<<EE / 32, 256, 0, stream>>>(p);
    }

    // ---- stress head ----
    {
        RT p{};
        for (int i = 0; i < 6; ++i) { p.W[i] = sW[i]; p.b[i] = sB[i]; }
        p.Xin = hbuf; p.atomCode = atomc;
        p.src = src; p.dst = dst; p.dist = dist; p.direction = dir;
        p.out = outS;
        k_mlp<4><<<EE / 32, 256, 0, stream>>>(p);
    }
}
